// Model_72258529787932
// MI455X (gfx1250) — matrix operand order repaired and checked by reading, not compiled, not run on hardware
//
#include <hip/hip_runtime.h>
#include <hip/hip_bf16.h>

// Problem constants (match reference)
#define NN    50000
#define EE    800000
#define INFD  256
#define HH    64
#define BB    4096
#define OUTN  2
#define FDIM  704          // 3H + IN + H + 3H = 192+256+64+192
#define EPSC  0.3f
#define NEGC  0.3f

typedef __attribute__((ext_vector_type(16))) __bf16 v16bf;
typedef __attribute__((ext_vector_type(8)))  float  v8f;

__device__ __forceinline__ float leaky_act(float v) { return v >= 0.f ? v : NEGC * v; }

// ---------------- degree normalization ----------------
__global__ void deg_count_kernel(const int* __restrict__ dst, float* __restrict__ deg, int E) {
    int i = blockIdx.x * blockDim.x + threadIdx.x;
    if (i < E) atomicAdd(&deg[dst[i]], 1.0f);
}

__global__ void deg_finalize_kernel(float* __restrict__ d, int n) {
    int i = blockIdx.x * blockDim.x + threadIdx.x;
    if (i < n) { float v = d[i]; d[i] = (v > 0.f) ? rsqrtf(v) : 0.f; }
}

// ---------------- per-node gate projections: pd = X.gw[:D], ps = X.gw[D:] ----------------
__global__ void gate_proj_kernel(const float* __restrict__ X, int D,
                                 const float* __restrict__ gw,
                                 float* __restrict__ pd, float* __restrict__ ps, int n) {
    int w    = (int)((blockIdx.x * (size_t)blockDim.x + threadIdx.x) >> 5);
    int lane = threadIdx.x & 31;
    if (w >= n) return;
    const float* xr = X + (size_t)w * D;
    float ad = 0.f, as = 0.f;
    for (int f = lane; f < D; f += 32) {
        float xv = xr[f];
        ad += xv * gw[f];
        as += xv * gw[D + f];
    }
    #pragma unroll
    for (int o = 16; o > 0; o >>= 1) {
        ad += __shfl_down(ad, o, 32);
        as += __shfl_down(as, o, 32);
    }
    if (lane == 0) { pd[w] = ad; ps[w] = as; }
}

// ---------------- per-edge gate + scatter-add ----------------
__global__ void edge_scatter_kernel(const float* __restrict__ X, int D,
                                    const int* __restrict__ src, const int* __restrict__ dst,
                                    const float* __restrict__ pd, const float* __restrict__ ps,
                                    const float* __restrict__ gb, const float* __restrict__ dn,
                                    float* __restrict__ z, int E) {
    int e    = (int)((blockIdx.x * (size_t)blockDim.x + threadIdx.x) >> 5);
    int lane = threadIdx.x & 31;
    if (e >= E) return;
    int s = src[e], t = dst[e];
    float a    = tanhf(pd[t] + ps[s] + gb[0]);
    float coef = a * dn[t] * dn[s];
    const float* xs = X + (size_t)s * D;
    float*       zt = z + (size_t)t * D;
    for (int f = lane; f < D; f += 32) atomicAdd(&zt[f], xs[f] * coef);
}

// ---------------- register-blocked WMMA GEMM -----------------------------------------
// C[:, colOfs+*] = leaky( (alpha*A + Z) @ op(W) + bias )
// Each wave computes RT row tiles (16*RT rows) x one 16-col tile; B fragment loaded once
// per K-step and reused across RT back-to-back v_wmma_f32_16x16x32_bf16 ops.
template<bool HAS_Z, bool WTRANS, bool HAS_BIAS, int RT>
__global__ void __launch_bounds__(32)
wmma_gemm_tpl(const float* __restrict__ A, const float* __restrict__ Z, float alpha,
              const float* __restrict__ W, int ldw,
              const float* __restrict__ bias,
              float* __restrict__ C, int K, int lda, int ldc, int colOfs) {
    int lane = threadIdx.x & 31;
    int m    = lane & 15;
    int half = lane >> 4;
    int ocol = blockIdx.y * 16 + m;                         // N index for B/C frags
    size_t rowTile = (size_t)blockIdx.x * (16 * RT);

    const float* arow[RT];
    const float* zrow[RT];
    #pragma unroll
    for (int t = 0; t < RT; ++t) {
        arow[t] = A + (rowTile + 16 * t + m) * lda;
        if (HAS_Z) zrow[t] = Z + (rowTile + 16 * t + m) * lda;
    }
    const float* wp = WTRANS ? (W + (size_t)ocol * ldw) : (W + ocol);

    v8f zero = {};
    v8f acc[RT];
    #pragma unroll
    for (int t = 0; t < RT; ++t) acc[t] = zero;

    for (int kb = 0; kb < K; kb += 32) {
        // B fragment (contiguous k order, same as A): K = 16*half + 2*j, N = lane&15
        v16bf bfrag;
        #pragma unroll
        for (int j = 0; j < 8; ++j) {
            int kk = kb + (half << 4) + (j << 1);
            float b0, b1;
            if (WTRANS) { b0 = wp[kk];                b1 = wp[kk + 1]; }
            else        { b0 = wp[(size_t)kk * ldw];  b1 = wp[(size_t)(kk + 1) * ldw]; }
            bfrag[2 * j]     = (__bf16)b0;
            bfrag[2 * j + 1] = (__bf16)b1;
        }
        #pragma unroll
        for (int t = 0; t < RT; ++t) {
            // A fragment (contiguous k order, same as B): K = 16*half + 2*j
            v16bf afrag;
            #pragma unroll
            for (int j = 0; j < 8; ++j) {
                int ka = kb + (half << 4) + (j << 1);
                float a0 = arow[t][ka], a1 = arow[t][ka + 1];
                if (HAS_Z) { a0 = alpha * a0 + zrow[t][ka]; a1 = alpha * a1 + zrow[t][ka + 1]; }
                afrag[2 * j]     = (__bf16)a0;
                afrag[2 * j + 1] = (__bf16)a1;
            }
            acc[t] = __builtin_amdgcn_wmma_f32_16x16x32_bf16(
                false, afrag, false, bfrag, (short)0, acc[t], false, false);
        }
    }

    float bv = HAS_BIAS ? bias[ocol] : 0.f;
    #pragma unroll
    for (int t = 0; t < RT; ++t) {
        #pragma unroll
        for (int r = 0; r < 8; ++r) {
            size_t row = rowTile + 16 * t + r + (half << 3);   // C layout: M = r + 8*half
            float v = acc[t][r] + bv;
            C[row * ldc + colOfs + ocol] = leaky_act(v);
        }
    }
}

// ---------------- gather selected rows into nf = [h2cat | h | x | raw2] ----------------
__global__ void gather_final_kernel(const float* __restrict__ h2cat, const float* __restrict__ h,
                                    const float* __restrict__ x, const float* __restrict__ raw2,
                                    const int* __restrict__ nodes, float* __restrict__ nf) {
    int b = blockIdx.x;
    int t = threadIdx.x;
    int node = nodes[b];
    float* o = nf + (size_t)b * FDIM;
    for (int i = t; i < 192; i += blockDim.x) o[i]       = h2cat[(size_t)node * 192 + i];
    for (int i = t; i < 256; i += blockDim.x) o[192 + i] = h[(size_t)node * 256 + i];
    for (int i = t; i < 64;  i += blockDim.x) o[448 + i] = x[(size_t)node * 64 + i];
    for (int i = t; i < 192; i += blockDim.x) o[512 + i] = raw2[(size_t)node * 192 + i];
}

// ---------------- final head: scores = s64 @ t3_w^T + t3_b -----------------------------
__global__ void head_kernel(const float* __restrict__ s64, const float* __restrict__ t3w,
                            const float* __restrict__ t3b, float* __restrict__ scores, int B) {
    int b = blockIdx.x * blockDim.x + threadIdx.x;
    if (b >= B) return;
    float a0 = t3b[0], a1 = t3b[1];
    const float* s = s64 + (size_t)b * 64;
    #pragma unroll 8
    for (int k = 0; k < 64; ++k) {
        float v = s[k];
        a0 += v * t3w[k];
        a1 += v * t3w[64 + k];
    }
    scores[2 * b]     = a0;
    scores[2 * b + 1] = a1;
}

extern "C" void kernel_launch(void* const* d_in, const int* in_sizes, int n_in,
                              void* d_out, int out_size, void* d_ws, size_t ws_size,
                              hipStream_t stream) {
    (void)in_sizes; (void)n_in; (void)out_size; (void)ws_size;

    const float* h     = (const float*)d_in[0];
    const int*   src[3] = {(const int*)d_in[1], (const int*)d_in[3], (const int*)d_in[5]};
    const int*   dst[3] = {(const int*)d_in[2], (const int*)d_in[4], (const int*)d_in[6]};
    const int*   nodes = (const int*)d_in[7];
    const float* t1_w  = (const float*)d_in[8];
    const float* t1_b  = (const float*)d_in[9];
    const float *gw1[3], *gb1[3], *hw1[3], *gw2[3], *gb2[3], *hw2[3];
    for (int i = 0; i < 3; ++i) {
        gw1[i] = (const float*)d_in[10 + 6 * i];
        gb1[i] = (const float*)d_in[11 + 6 * i];
        hw1[i] = (const float*)d_in[12 + 6 * i];
        gw2[i] = (const float*)d_in[13 + 6 * i];
        gb2[i] = (const float*)d_in[14 + 6 * i];
        hw2[i] = (const float*)d_in[15 + 6 * i];
    }
    const float* t2_w = (const float*)d_in[28];
    const float* t2_b = (const float*)d_in[29];
    const float* t3_w = (const float*)d_in[30];
    const float* t3_b = (const float*)d_in[31];

    // workspace carve-up (floats)
    float* ws    = (float*)d_ws;
    float* dnorm = ws; ws += 3 * (size_t)NN;        // d1 | d2 | d3
    float* x     = ws; ws += (size_t)NN * HH;       // [N,64]
    float* z1    = ws; ws += (size_t)NN * HH;       // [N,64]
    float* raw2  = ws; ws += (size_t)NN * 3 * HH;   // [N,192]
    float* z2    = ws; ws += (size_t)NN * 3 * HH;   // [N,192]
    float* h2cat = ws; ws += (size_t)NN * 3 * HH;   // [N,192]
    float* pd    = ws; ws += (size_t)NN;
    float* psb   = ws; ws += (size_t)NN;
    float* nf    = ws; ws += (size_t)BB * FDIM;     // [B,704]

    float* scores = (float*)d_out;                          // [B,2]
    float* s64    = (float*)d_out + (size_t)BB * OUTN;      // [B,64]

    // 1) degree norms for the 3 edge sets
    hipMemsetAsync(dnorm, 0, 3 * (size_t)NN * sizeof(float), stream);
    for (int k = 0; k < 3; ++k)
        deg_count_kernel<<<(EE + 255) / 256, 256, 0, stream>>>(dst[k], dnorm + (size_t)k * NN, EE);
    deg_finalize_kernel<<<(3 * NN + 255) / 256, 256, 0, stream>>>(dnorm, 3 * NN);

    // 2) x = leaky(h @ t1_w^T + t1_b)   [N,256] x [256,64]; 625 blocks x 80 rows
    wmma_gemm_tpl<false, true, true, 5><<<dim3(NN / 80, HH / 16), 32, 0, stream>>>(
        h, nullptr, 1.0f, t1_w, /*ldw=*/INFD, t1_b,
        x, /*K=*/INFD, /*lda=*/INFD, /*ldc=*/HH, /*colOfs=*/0);

    // 3) layer 1: three edge sets -> raw2[:, k*64 : k*64+64]
    for (int k = 0; k < 3; ++k) {
        gate_proj_kernel<<<(NN + 7) / 8, 256, 0, stream>>>(x, HH, gw1[k], pd, psb, NN);
        hipMemsetAsync(z1, 0, (size_t)NN * HH * sizeof(float), stream);
        edge_scatter_kernel<<<(EE + 7) / 8, 256, 0, stream>>>(
            x, HH, src[k], dst[k], pd, psb, gb1[k], dnorm + (size_t)k * NN, z1, EE);
        wmma_gemm_tpl<true, false, false, 5><<<dim3(NN / 80, HH / 16), 32, 0, stream>>>(
            x, z1, EPSC, hw1[k], /*ldw=*/HH, nullptr,
            raw2, /*K=*/HH, /*lda=*/HH, /*ldc=*/3 * HH, /*colOfs=*/k * HH);
    }

    // 4) layer 2: three edge sets -> h2cat[:, k*64 : k*64+64]
    for (int k = 0; k < 3; ++k) {
        gate_proj_kernel<<<(NN + 7) / 8, 256, 0, stream>>>(raw2, 3 * HH, gw2[k], pd, psb, NN);
        hipMemsetAsync(z2, 0, (size_t)NN * 3 * HH * sizeof(float), stream);
        edge_scatter_kernel<<<(EE + 7) / 8, 256, 0, stream>>>(
            raw2, 3 * HH, src[k], dst[k], pd, psb, gb2[k], dnorm + (size_t)k * NN, z2, EE);
        wmma_gemm_tpl<true, false, false, 5><<<dim3(NN / 80, HH / 16), 32, 0, stream>>>(
            raw2, z2, EPSC, hw2[k], /*ldw=*/HH, nullptr,
            h2cat, /*K=*/3 * HH, /*lda=*/3 * HH, /*ldc=*/3 * HH, /*colOfs=*/k * HH);
    }

    // 5) gather selected nodes into nf = [h2_1|h2_2|h2_3 | raw0 | raw1 | raw2]  [B,704]
    gather_final_kernel<<<BB, 128, 0, stream>>>(h2cat, h, x, raw2, nodes, nf);

    // 6) s64 = leaky(nf @ t2_w^T + t2_b)   [4096,704] x [704,64]; 64 blocks x 64 rows
    wmma_gemm_tpl<false, true, true, 4><<<dim3(BB / 64, HH / 16), 32, 0, stream>>>(
        nf, nullptr, 1.0f, t2_w, /*ldw=*/FDIM, t2_b,
        s64, /*K=*/FDIM, /*lda=*/FDIM, /*ldc=*/HH, /*colOfs=*/0);

    // 7) scores = s64 @ t3_w^T + t3_b   [4096,64] x [64,2]
    head_kernel<<<(BB + 255) / 256, 256, 0, stream>>>(s64, t3_w, t3_b, scores, BB);
}
